// Model_26817775796469
// MI455X (gfx1250) — matrix operand order repaired and checked by reading, not compiled, not run on hardware
//
#include <hip/hip_runtime.h>
#include <hip/hip_bf16.h>

#define EPSV 1e-4f

typedef __attribute__((ext_vector_type(16))) _Float16 v16h;
typedef __attribute__((ext_vector_type(8)))  _Float16 v8h;
typedef __attribute__((ext_vector_type(8)))  float    v8f;

__device__ __forceinline__ v16h cat16(v8h lo, v8h hi) {
  v16h r;
#pragma unroll
  for (int i = 0; i < 8; ++i) { r[i] = lo[i]; r[i + 8] = hi[i]; }
  return r;
}

__device__ __forceinline__ _Float16 signh(float t) {
  return (_Float16)((t > 0.f) ? 1.f : ((t < 0.f) ? -1.f : 0.f));
}

// ---------------------------------------------------------------------------
// Weight pre-transforms: fp32 OIHW -> f16 [tap][co][ci]
// ---------------------------------------------------------------------------
__global__ void prep_w_kernel(const float* __restrict__ w, _Float16* __restrict__ wt,
                              int COUT, int CIN, int KS) {
  int idx = blockIdx.x * blockDim.x + threadIdx.x;
  int total = KS * KS * COUT * CIN;
  if (idx >= total) return;
  int ci = idx % CIN;
  int t  = idx / CIN;
  int co = t % COUT;
  int tap = t / COUT;
  int ky = tap / KS, kx = tap % KS;
  wt[idx] = (_Float16)w[((long)(co * CIN + ci) * KS + ky) * KS + kx];
}

// conv1 weights: [co][96], k = tap*3 + ci (tap = dy*5+dx), zero-padded past 75
__global__ void prep_w1_kernel(const float* __restrict__ w, _Float16* __restrict__ wt) {
  int idx = blockIdx.x * blockDim.x + threadIdx.x;
  if (idx >= 192 * 96) return;
  int k = idx % 96;
  int co = idx / 96;
  float val = 0.f;
  if (k < 75) {
    int tap = k / 3, ci = k - tap * 3;
    int ky = tap / 5, kx = tap - ky * 5;
    val = w[((long)(co * 3 + ci) * 5 + ky) * 5 + kx];
  }
  wt[idx] = (_Float16)val;
}

// conv9 weights padded to Cout=16: [16][192]
__global__ void prep_w9_kernel(const float* __restrict__ w, _Float16* __restrict__ wt) {
  int idx = blockIdx.x * blockDim.x + threadIdx.x;
  if (idx >= 16 * 192) return;
  int ci = idx % 192;
  int co = idx / 192;
  wt[idx] = (_Float16)((co < 10) ? w[co * 192 + ci] : 0.f);
}

// ---------------------------------------------------------------------------
// Input binarization: x (NCHW f32) -> A1 (NHWC f16, sign(BN1(x)))
// ---------------------------------------------------------------------------
__global__ void binact_input(const float* __restrict__ x,
                             const float* __restrict__ g, const float* __restrict__ be,
                             const float* __restrict__ m, const float* __restrict__ v,
                             _Float16* __restrict__ out) {
  int idx = blockIdx.x * blockDim.x + threadIdx.x;
  const int TOT = 256 * 32 * 32 * 3;
  if (idx >= TOT) return;
  int c = idx % 3;
  int t = idx / 3;
  int w = t & 31; t >>= 5;
  int h = t & 31;
  int n = t >> 5;
  float xv = x[((long)((n * 3 + c) * 32 + h)) * 32 + w];
  float tt = (xv - m[c]) * rsqrtf(v[c] + EPSV) * g[c] + be[c];
  out[idx] = signh(tt);
}

// ---------------------------------------------------------------------------
// conv1: 5x5, 3->192 via WMMA with packed K=96 (tap*3+ci).
// NW waves per block share the 37 KB weight block staged in LDS.
// Epilogue: ReLU -> BN_a(no affine) -> ReLU -> BN2(affine) -> sign
// ---------------------------------------------------------------------------
template <int NW>
__global__ __launch_bounds__(NW * 32) void conv1_wmma(
    const _Float16* __restrict__ a1, const _Float16* __restrict__ wt1,
    const float* __restrict__ b1,
    const float* __restrict__ ma, const float* __restrict__ va,
    const float* __restrict__ g2, const float* __restrict__ be2,
    const float* __restrict__ m2, const float* __restrict__ v2,
    _Float16* __restrict__ out) {
  __shared__ _Float16 sW[192 * 96];
  const int tid = threadIdx.x;
  {
    const v8h* g = (const v8h*)wt1;
    v8h* s = (v8h*)sW;
    for (int i = tid; i < 192 * 96 / 8; i += NW * 32) s[i] = g[i];
  }
  __syncthreads();

  const int wave = tid >> 5;
  const int lane = tid & 31;
  const int col = lane & 15;
  const int hs = lane >> 4;
  const long p = ((long)blockIdx.x * NW + wave) * 16 + col;
  const int w0 = (int)(p & 31);
  const int h0 = (int)((p >> 5) & 31);
  const int n  = (int)(p >> 10);

  // Build 3 B-fragments by im2col gather (K packed: tap*3+ci, padded to 96)
  v16h B[3];
#pragma unroll
  for (int c3 = 0; c3 < 3; ++c3) {
    v8h blo = {}, bhi = {};
#pragma unroll
    for (int j = 0; j < 16; ++j) {
      int k = c3 * 32 + hs * 16 + j;
      _Float16 val = (_Float16)0.f;
      if (k < 75) {
        int tap = k / 3, ci = k - tap * 3;
        int dy = tap / 5, dx = tap - dy * 5;
        int h2 = h0 + dy - 2, w2 = w0 + dx - 2;
        if ((unsigned)h2 < 32u && (unsigned)w2 < 32u)
          val = a1[((long)((n * 32 + h2) * 32 + w2)) * 3 + ci];
      }
      if (j < 8) blo[j] = val; else bhi[j - 8] = val;
    }
    B[c3] = cat16(blo, bhi);
  }

  v8f acc[12] = {};
#pragma unroll
  for (int t = 0; t < 12; ++t) {
#pragma unroll
    for (int c3 = 0; c3 < 3; ++c3) {
      const _Float16* ap = sW + (t * 16 + col) * 96 + c3 * 32 + hs * 16;
      v16h A = cat16(*(const v8h*)ap, *(const v8h*)(ap + 8));
      acc[t] = __builtin_amdgcn_wmma_f32_16x16x32_f16(false, A, false, B[c3],
                                                      (short)0, acc[t], false, false);
    }
  }

#pragma unroll
  for (int t = 0; t < 12; ++t) {
    v8h o;
#pragma unroll
    for (int r = 0; r < 8; ++r) {
      int co = t * 16 + hs * 8 + r;
      float y = fmaxf(acc[t][r] + b1[co], 0.f);
      y = fmaxf((y - ma[co]) * rsqrtf(va[co] + EPSV), 0.f);
      float s = (y - m2[co]) * rsqrtf(v2[co] + EPSV) * g2[co] + be2[co];
      o[r] = signh(s);
    }
    *(v8h*)(out + p * 192 + t * 16 + hs * 8) = o;
  }
}

// ---------------------------------------------------------------------------
// Generic binary conv as implicit GEMM with WMMA + LDS-staged weights.
//   act: NHWC f16 [B,H,W,CIN] (binarized), wt: [KS*KS][COUT][CIN] f16
//   NW waves per block, each owning a 16-position tile; the per-tap weight
//   slice (COUT x CIN f16) is staged cooperatively into LDS and read back
//   as A-fragments via ds_load_b128, amortizing weight traffic NW-fold.
// EPI: 0 = ReLU->BN(affine)->sign    (write next layer's binarized input)
//      1 = ReLU -> f16               (write pooling input)
//      2 = ReLU->BN(no affine, b2)->BN(affine)->sign   (conv8 path)
// ---------------------------------------------------------------------------
template <int CIN, int COUT, int KS, int PAD, int H, int W, int EPI, int NW>
__global__ __launch_bounds__(NW * 32) void conv_bin_wmma(
    const _Float16* __restrict__ act, const _Float16* __restrict__ wt,
    const float* __restrict__ bias,
    const float* __restrict__ bg, const float* __restrict__ bb,
    const float* __restrict__ bm, const float* __restrict__ bv,
    const float* __restrict__ b2m, const float* __restrict__ b2v,
    _Float16* __restrict__ out) {
  constexpr int CT = COUT / 16;
  __shared__ _Float16 sW[COUT * CIN];

  const int tid = threadIdx.x;
  const int wave = tid >> 5;
  const int lane = tid & 31;
  const int col = lane & 15;
  const int hs = lane >> 4;
  const long p = ((long)blockIdx.x * NW + wave) * 16 + col;
  const int w0 = (int)(p % W);
  const int h0 = (int)((p / W) % H);
  const int n  = (int)(p / ((long)W * H));

  v8f acc[CT] = {};

  for (int tap = 0; tap < KS * KS; ++tap) {
    // Cooperative stage of this tap's weight slice into LDS.
    if (KS > 1 && tap > 0) __syncthreads();  // protect previous tap's reads
    {
      const v8h* g = (const v8h*)(wt + (long)tap * COUT * CIN);
      v8h* s = (v8h*)sW;
      for (int i = tid; i < COUT * CIN / 8; i += NW * 32) s[i] = g[i];
    }
    __syncthreads();

    const int dy = tap / KS, dx = tap % KS;
    const int h2 = h0 + dy - PAD, w2 = w0 + dx - PAD;
    const bool valid = ((unsigned)h2 < (unsigned)H) && ((unsigned)w2 < (unsigned)W);
    for (int cc = 0; cc < CIN; cc += 32) {
      v8h blo = {}, bhi = {};
      if (valid) {
        const _Float16* bp = act + ((long)((n * H + h2) * W + w2)) * CIN + cc + hs * 16;
        blo = *(const v8h*)bp;
        bhi = *(const v8h*)(bp + 8);
      }
      const v16h B = cat16(blo, bhi);
#pragma unroll
      for (int t = 0; t < CT; ++t) {
        const _Float16* ap = sW + (t * 16 + col) * CIN + cc + hs * 16;
        v16h A = cat16(*(const v8h*)ap, *(const v8h*)(ap + 8));
        acc[t] = __builtin_amdgcn_wmma_f32_16x16x32_f16(false, A, false, B,
                                                        (short)0, acc[t], false, false);
      }
    }
  }

#pragma unroll
  for (int t = 0; t < CT; ++t) {
    v8h o;
#pragma unroll
    for (int r = 0; r < 8; ++r) {
      const int co = t * 16 + hs * 8 + r;
      float y = fmaxf(acc[t][r] + bias[co], 0.f);
      if constexpr (EPI == 1) {
        o[r] = (_Float16)y;
      } else {
        if constexpr (EPI == 2) y = (y - b2m[co]) * rsqrtf(b2v[co] + EPSV);
        float s = (y - bm[co]) * rsqrtf(bv[co] + EPSV) * bg[co] + bb[co];
        o[r] = signh(s);
      }
    }
    *(v8h*)(out + p * COUT + t * 16 + hs * 8) = o;
  }
}

// ---------------------------------------------------------------------------
// conv7: 3x3, 192->192 on 8x8, whole per-image slice staged in LDS (24 KB).
// Epilogue: ReLU -> BN8 -> sign
// ---------------------------------------------------------------------------
__global__ __launch_bounds__(128) void conv7_wmma_lds(
    const _Float16* __restrict__ act, const _Float16* __restrict__ wt,
    const float* __restrict__ bias,
    const float* __restrict__ bg, const float* __restrict__ bb,
    const float* __restrict__ bm, const float* __restrict__ bv,
    _Float16* __restrict__ out) {
  __shared__ _Float16 sA[64 * 192];
  const int n = blockIdx.x;
  const int tid = threadIdx.x;
  {
    const v8h* g = (const v8h*)(act + (long)n * 64 * 192);
    v8h* s = (v8h*)sA;
#pragma unroll
    for (int i = 0; i < 12; ++i) s[tid + i * 128] = g[tid + i * 128];  // 1536 x 16B
  }
  __syncthreads();

  const int wave = tid >> 5;       // 0..3 -> position tile
  const int lane = tid & 31;
  const int col = lane & 15, hs = lane >> 4;
  const int pl = wave * 16 + col;  // 0..63
  const int h0 = pl >> 3, w0 = pl & 7;

  v8f acc[12] = {};
  for (int tap = 0; tap < 9; ++tap) {
    int dy = tap / 3, dx = tap % 3;
    int h2 = h0 + dy - 1, w2 = w0 + dx - 1;
    bool valid = ((unsigned)h2 < 8u) && ((unsigned)w2 < 8u);
    const _Float16* wtap = wt + (long)tap * 192 * 192;
    for (int cc = 0; cc < 192; cc += 32) {
      v8h blo = {}, bhi = {};
      if (valid) {
        const _Float16* bp = sA + (long)(h2 * 8 + w2) * 192 + cc + hs * 16;
        blo = *(const v8h*)bp;
        bhi = *(const v8h*)(bp + 8);
      }
      v16h B = cat16(blo, bhi);
#pragma unroll
      for (int t = 0; t < 12; ++t) {
        const _Float16* ap = wtap + (long)(t * 16 + col) * 192 + cc + hs * 16;
        v16h A = cat16(*(const v8h*)ap, *(const v8h*)(ap + 8));
        acc[t] = __builtin_amdgcn_wmma_f32_16x16x32_f16(false, A, false, B,
                                                        (short)0, acc[t], false, false);
      }
    }
  }

  long p = (long)n * 64 + pl;
#pragma unroll
  for (int t = 0; t < 12; ++t) {
    v8h o;
#pragma unroll
    for (int r = 0; r < 8; ++r) {
      int co = t * 16 + hs * 8 + r;
      float y = fmaxf(acc[t][r] + bias[co], 0.f);
      float s = (y - bm[co]) * rsqrtf(bv[co] + EPSV) * bg[co] + bb[co];
      o[r] = signh(s);
    }
    *(v8h*)(out + p * 192 + t * 16 + hs * 8) = o;
  }
}

// ---------------------------------------------------------------------------
// conv9: 1x1 192->10 (padded to 16) + ReLU + spatial mean -> out[256,10] f32
// One wave per image; lane-shuffle reduction over the 64 positions.
// ---------------------------------------------------------------------------
__global__ __launch_bounds__(32) void conv9_wmma_mean(
    const _Float16* __restrict__ act, const _Float16* __restrict__ wt,
    const float* __restrict__ bias, float* __restrict__ out) {
  const int n = blockIdx.x;
  const int lane = threadIdx.x, col = lane & 15, hs = lane >> 4;
  float ps[8] = {0.f, 0.f, 0.f, 0.f, 0.f, 0.f, 0.f, 0.f};
  for (int pt = 0; pt < 4; ++pt) {
    const _Float16* arow = act + ((long)n * 64 + pt * 16 + col) * 192;
    v8f acc = {};
    for (int cc = 0; cc < 192; cc += 32) {
      const _Float16* bp = arow + cc + hs * 16;
      v16h B = cat16(*(const v8h*)bp, *(const v8h*)(bp + 8));
      const _Float16* ap = wt + (long)col * 192 + cc + hs * 16;
      v16h A = cat16(*(const v8h*)ap, *(const v8h*)(ap + 8));
      acc = __builtin_amdgcn_wmma_f32_16x16x32_f16(false, A, false, B,
                                                   (short)0, acc, false, false);
    }
#pragma unroll
    for (int r = 0; r < 8; ++r) {
      int co = hs * 8 + r;
      float bi = (co < 10) ? bias[co] : 0.f;
      ps[r] += fmaxf(acc[r] + bi, 0.f);
    }
  }
#pragma unroll
  for (int r = 0; r < 8; ++r) {
    float s = ps[r];
    s += __shfl_xor(s, 1, 32);
    s += __shfl_xor(s, 2, 32);
    s += __shfl_xor(s, 4, 32);
    s += __shfl_xor(s, 8, 32);
    if (col == 0) {
      int co = hs * 8 + r;
      if (co < 10) out[n * 10 + co] = s * (1.f / 64.f);
    }
  }
}

// ---------------------------------------------------------------------------
// Pools (NHWC f16), fused with next layer's BN + sign
// ---------------------------------------------------------------------------
__global__ void maxpool_bnsign(const _Float16* __restrict__ in,  // [256,32,32,96]
                               const float* __restrict__ g, const float* __restrict__ be,
                               const float* __restrict__ m, const float* __restrict__ v,
                               _Float16* __restrict__ out) {     // [256,16,16,96]
  int idx = blockIdx.x * blockDim.x + threadIdx.x;
  const int TOT = 256 * 16 * 16 * 96;
  if (idx >= TOT) return;
  int c = idx % 96;
  int t = idx / 96;
  int wo = t & 15; t >>= 4;
  int ho = t & 15;
  int n = t >> 4;
  float mx = -3.0e38f;
  for (int dy = 0; dy < 3; ++dy) {
    int h = ho * 2 - 1 + dy;
    if ((unsigned)h >= 32u) continue;
    for (int dx = 0; dx < 3; ++dx) {
      int w = wo * 2 - 1 + dx;
      if ((unsigned)w >= 32u) continue;
      float val = (float)in[((long)((n * 32 + h) * 32 + w)) * 96 + c];
      mx = fmaxf(mx, val);
    }
  }
  float tt = (mx - m[c]) * rsqrtf(v[c] + EPSV) * g[c] + be[c];
  out[idx] = signh(tt);
}

__global__ void avgpool_bnsign(const _Float16* __restrict__ in,  // [256,16,16,192]
                               const float* __restrict__ g, const float* __restrict__ be,
                               const float* __restrict__ m, const float* __restrict__ v,
                               _Float16* __restrict__ out) {     // [256,8,8,192]
  int idx = blockIdx.x * blockDim.x + threadIdx.x;
  const int TOT = 256 * 8 * 8 * 192;
  if (idx >= TOT) return;
  int c = idx % 192;
  int t = idx / 192;
  int wo = t & 7; t >>= 3;
  int ho = t & 7;
  int n = t >> 3;
  float s = 0.f;
  for (int dy = 0; dy < 3; ++dy) {
    int h = ho * 2 - 1 + dy;
    if ((unsigned)h >= 16u) continue;
    for (int dx = 0; dx < 3; ++dx) {
      int w = wo * 2 - 1 + dx;
      if ((unsigned)w >= 16u) continue;
      s += (float)in[((long)((n * 16 + h) * 16 + w)) * 192 + c];
    }
  }
  s *= (1.f / 9.f);
  float tt = (s - m[c]) * rsqrtf(v[c] + EPSV) * g[c] + be[c];
  out[idx] = signh(tt);
}

// ---------------------------------------------------------------------------
// Launch
// ---------------------------------------------------------------------------
extern "C" void kernel_launch(void* const* d_in, const int* in_sizes, int n_in,
                              void* d_out, int out_size, void* d_ws, size_t ws_size,
                              hipStream_t stream) {
  (void)in_sizes; (void)n_in; (void)out_size; (void)ws_size;
  const float* X = (const float*)d_in[0];
  auto G  = [&](int i) { return (const float*)d_in[1 + (i - 1) * 6 + 0]; };
  auto BE = [&](int i) { return (const float*)d_in[1 + (i - 1) * 6 + 1]; };
  auto Mn = [&](int i) { return (const float*)d_in[1 + (i - 1) * 6 + 2]; };
  auto Vr = [&](int i) { return (const float*)d_in[1 + (i - 1) * 6 + 3]; };
  auto Wt = [&](int i) { return (const float*)d_in[1 + (i - 1) * 6 + 4]; };
  auto Bi = [&](int i) { return (const float*)d_in[1 + (i - 1) * 6 + 5]; };
  const float* ma = (const float*)d_in[55];
  const float* va = (const float*)d_in[56];
  const float* mb = (const float*)d_in[57];
  const float* vb = (const float*)d_in[58];

  char* base = (char*)d_ws;
  size_t off = 0;
  auto alloc = [&](size_t bytes) -> char* {
    char* p = base + off;
    off += (bytes + 255) & ~(size_t)255;
    return p;
  };
  _Float16* wt1 = (_Float16*)alloc((size_t)192 * 96 * 2);
  _Float16* wt2 = (_Float16*)alloc((size_t)160 * 192 * 2);
  _Float16* wt3 = (_Float16*)alloc((size_t)96 * 160 * 2);
  _Float16* wt4 = (_Float16*)alloc((size_t)25 * 192 * 96 * 2);
  _Float16* wt5 = (_Float16*)alloc((size_t)192 * 192 * 2);
  _Float16* wt6 = (_Float16*)alloc((size_t)192 * 192 * 2);
  _Float16* wt7 = (_Float16*)alloc((size_t)9 * 192 * 192 * 2);
  _Float16* wt8 = (_Float16*)alloc((size_t)192 * 192 * 2);
  _Float16* wt9 = (_Float16*)alloc((size_t)16 * 192 * 2);
  _Float16* A1   = (_Float16*)alloc((size_t)256 * 1024 * 3 * 2);
  _Float16* bufX = (_Float16*)alloc((size_t)256 * 1024 * 192 * 2);  // 100.7 MB
  _Float16* bufY = (_Float16*)alloc((size_t)256 * 1024 * 160 * 2);  // 83.9 MB

  // Weight transforms
  prep_w1_kernel<<<(192 * 96 + 255) / 256, 256, 0, stream>>>(Wt(1), wt1);
  auto prep = [&](const float* w, _Float16* wt, int CO, int CI, int KS) {
    int tot = KS * KS * CO * CI;
    prep_w_kernel<<<(tot + 255) / 256, 256, 0, stream>>>(w, wt, CO, CI, KS);
  };
  prep(Wt(2), wt2, 160, 192, 1);
  prep(Wt(3), wt3, 96, 160, 1);
  prep(Wt(4), wt4, 192, 96, 5);
  prep(Wt(5), wt5, 192, 192, 1);
  prep(Wt(6), wt6, 192, 192, 1);
  prep(Wt(7), wt7, 192, 192, 3);
  prep(Wt(8), wt8, 192, 192, 1);
  prep_w9_kernel<<<(16 * 192 + 255) / 256, 256, 0, stream>>>(Wt(9), wt9);

  // Input binarization
  binact_input<<<(256 * 32 * 32 * 3 + 255) / 256, 256, 0, stream>>>(
      X, G(1), BE(1), Mn(1), Vr(1), A1);

  // conv1: A1 -> bufX (A2); 8 waves share LDS-staged weights
  conv1_wmma<8><<<2048, 256, 0, stream>>>(A1, wt1, Bi(1), ma, va,
                                          G(2), BE(2), Mn(2), Vr(2), bufX);
  // conv2: bufX -> bufY (A3), BN3+sign
  conv_bin_wmma<192, 160, 1, 0, 32, 32, 0, 8><<<2048, 256, 0, stream>>>(
      bufX, wt2, Bi(2), G(3), BE(3), Mn(3), Vr(3),
      (const float*)nullptr, (const float*)nullptr, bufY);
  // conv3: bufY -> bufX (P3), ReLU f16
  conv_bin_wmma<160, 96, 1, 0, 32, 32, 1, 8><<<2048, 256, 0, stream>>>(
      bufY, wt3, Bi(3), (const float*)nullptr, (const float*)nullptr,
      (const float*)nullptr, (const float*)nullptr,
      (const float*)nullptr, (const float*)nullptr, bufX);
  // maxpool + BN4 + sign: bufX -> bufY (A4)
  maxpool_bnsign<<<(256 * 16 * 16 * 96 + 255) / 256, 256, 0, stream>>>(
      bufX, G(4), BE(4), Mn(4), Vr(4), bufY);
  // conv4: bufY -> bufX (A5), BN5+sign (per-tap weight slice staged in LDS)
  conv_bin_wmma<96, 192, 5, 2, 16, 16, 0, 8><<<512, 256, 0, stream>>>(
      bufY, wt4, Bi(4), G(5), BE(5), Mn(5), Vr(5),
      (const float*)nullptr, (const float*)nullptr, bufX);
  // conv5: bufX -> bufY (A6), BN6+sign
  conv_bin_wmma<192, 192, 1, 0, 16, 16, 0, 8><<<512, 256, 0, stream>>>(
      bufX, wt5, Bi(5), G(6), BE(6), Mn(6), Vr(6),
      (const float*)nullptr, (const float*)nullptr, bufY);
  // conv6: bufY -> bufX (P6), ReLU f16
  conv_bin_wmma<192, 192, 1, 0, 16, 16, 1, 8><<<512, 256, 0, stream>>>(
      bufY, wt6, Bi(6), (const float*)nullptr, (const float*)nullptr,
      (const float*)nullptr, (const float*)nullptr,
      (const float*)nullptr, (const float*)nullptr, bufX);
  // avgpool + BN7 + sign: bufX -> bufY (A7)
  avgpool_bnsign<<<(256 * 8 * 8 * 192 + 255) / 256, 256, 0, stream>>>(
      bufX, G(7), BE(7), Mn(7), Vr(7), bufY);
  // conv7 (activations staged in LDS): bufY -> bufX (A8), BN8+sign
  conv7_wmma_lds<<<256, 128, 0, stream>>>(bufY, wt7, Bi(7),
                                          G(8), BE(8), Mn(8), Vr(8), bufX);
  // conv8: bufX -> bufY (A9), ReLU -> BN_b -> BN9 -> sign
  conv_bin_wmma<192, 192, 1, 0, 8, 8, 2, 8><<<128, 256, 0, stream>>>(
      bufX, wt8, Bi(8), G(9), BE(9), Mn(9), Vr(9), mb, vb, bufY);
  // conv9 + spatial mean -> d_out [256,10] f32
  conv9_wmma_mean<<<256, 32, 0, stream>>>(bufY, wt9, Bi(9), (float*)d_out);
}
